// IGATLayer_20169166422637
// MI455X (gfx1250) — matrix operand order repaired and checked by reading, not compiled, not run on hardware
//
#include <hip/hip_runtime.h>
#include <hip/hip_bf16.h>

typedef __attribute__((ext_vector_type(2)))  float        v2f;
typedef __attribute__((ext_vector_type(8)))  float        v8f;
typedef __attribute__((ext_vector_type(16))) __bf16       v16bf;
typedef __attribute__((ext_vector_type(4)))  unsigned int v4u;
typedef __attribute__((ext_vector_type(8)))  int          v8i;
typedef __attribute__((ext_vector_type(4)))  int          v4i;

#define NN 8192
#define DD 256

#if defined(__has_builtin)
#if __has_builtin(__builtin_amdgcn_tensor_load_to_lds) && \
    __has_builtin(__builtin_amdgcn_s_wait_tensorcnt)
#define HAVE_TDM 1
#endif
#endif

#ifdef HAVE_TDM
#define TDM_WAIT() __builtin_amdgcn_s_wait_tensorcnt((short)0)
#else
#define TDM_WAIT()
#endif

// ---------------------------------------------------------------------------
// Kernel 0: zero the h' accumulator (atomically accumulated by attn_gemm).
// ---------------------------------------------------------------------------
__global__ void zero_kernel(float* __restrict__ p, int n) {
  int i = blockIdx.x * 256 + threadIdx.x;
  if (i < n) p[i] = 0.0f;
}

// ---------------------------------------------------------------------------
// Kernel 1: Wh = X @ W  via V_WMMA_F32_16X16X4_F32 (fp32, K=4 per op).
// ---------------------------------------------------------------------------
__global__ void wh_gemm(const float* __restrict__ X, const float* __restrict__ W,
                        float* __restrict__ Wh) {
  int lane = threadIdx.x & 31;
  int wv   = threadIdx.x >> 5;
  int tile = blockIdx.x * 8 + wv;     // 8192 tiles = 512 row-tiles x 16 col-tiles
  int it = tile >> 4, nt = tile & 15;
  int i0 = it * 16, n0 = nt * 16;
  int ml = lane & 15, hi = lane >> 4;

  v8f c = {};
  const float* xrow = X + (i0 + ml) * DD;
  const float* wcol = W + n0 + ml;
  for (int k0 = 0; k0 < DD; k0 += 4) {
    int k = k0 + 2 * hi;
    v2f a, b;
    a.x = xrow[k + 0];
    a.y = xrow[k + 1];
    b.x = wcol[(k + 0) * DD];
    b.y = wcol[(k + 1) * DD];
    c = __builtin_amdgcn_wmma_f32_16x16x4_f32(false, a, false, b,
                                              (short)0, c, false, false);
  }
  float* orow = Wh + n0 + ml;
#pragma unroll
  for (int r = 0; r < 8; ++r) orow[(i0 + r + 8 * hi) * DD] = c[r];
}

// ---------------------------------------------------------------------------
// Kernel 2: f_src = Wh@a_src, f_dst = Wh@a_dst. One wave per row.
// ---------------------------------------------------------------------------
__global__ void fvec_kernel(const float* __restrict__ Wh,
                            const float* __restrict__ a_src,
                            const float* __restrict__ a_dst,
                            float* __restrict__ fs, float* __restrict__ fd) {
  int lane = threadIdx.x & 31;
  int row  = (blockIdx.x * blockDim.x + threadIdx.x) >> 5;
  const float* r = Wh + row * DD;
  float s1 = 0.0f, s2 = 0.0f;
  for (int k = lane; k < DD; k += 32) {
    float v = r[k];
    s1 += v * a_src[k];
    s2 += v * a_dst[k];
  }
  for (int off = 16; off; off >>= 1) {
    s1 += __shfl_down(s1, off, 32);
    s2 += __shfl_down(s2, off, 32);
  }
  if (lane == 0) { fs[row] = s1; fd[row] = s2; }
}

// ---------------------------------------------------------------------------
// Kernel 3: repack Wh (fp32 row-major) -> bf16 in WMMA-B register-staged
// layout: dword ((jt*16+nt)*32 + lane)*8 + vg  holds  B[k=2(vg&3)+8(lane>>4)+16(vg>>2)..+1][n=nt*16+lane%16]
// of the 32x16 tile jt = j/32 (one contiguous 32B load per lane at consume).
// ---------------------------------------------------------------------------
__global__ void repack_kernel(const float* __restrict__ Wh,
                              unsigned int* __restrict__ staged) {
  int t    = blockIdx.x * 256 + threadIdx.x;   // 0 .. 1048575
  int vg   = t & 7;
  int lane = (t >> 3) & 31;
  int tl   = t >> 8;                 // jt*16 + nt
  int nt   = tl & 15, jt = tl >> 4;
  int j    = jt * 32 + 2 * (vg & 3) + 8 * (lane >> 4) + 16 * (vg >> 2);
  int n    = nt * 16 + (lane & 15);
  __bf16 b0 = (__bf16)Wh[j * DD + n];
  __bf16 b1 = (__bf16)Wh[(j + 1) * DD + n];
  unsigned int u0 = (unsigned int)__builtin_bit_cast(unsigned short, b0);
  unsigned int u1 = (unsigned int)__builtin_bit_cast(unsigned short, b1);
  staged[t] = u0 | (u1 << 16);
}

// ---------------------------------------------------------------------------
// Kernel 4: the only HBM-bound pass (268 MB adjacency). One block per row:
// ballot-pack adjacency to bits + online (max, sum-exp) softmax stats.
// ---------------------------------------------------------------------------
__global__ void rowstats_kernel(const int* __restrict__ adj,
                                const float* __restrict__ fs,
                                const float* __restrict__ fd,
                                float* __restrict__ Mrow, float* __restrict__ invZ,
                                unsigned int* __restrict__ packed) {
  __shared__ float sm[8], ss[8];
  int row  = blockIdx.x;
  int lane = threadIdx.x & 31, wv = threadIdx.x >> 5;
  float fsr = fs[row];
  float m = -3.0e38f, s = 0.0f;
  const int* arow = adj + (long long)row * NN;
  for (int it = 0; it < 32; ++it) {
    int widx = it * 8 + wv;
    int j = widx * 32 + lane;
    bool on = arow[j] > 0;
    unsigned int bits = (unsigned int)__ballot(on);
    if (lane == 0) packed[row * 256 + widx] = bits;
    if (on) {
      float e = fsr + fd[j];
      e = e > 0.0f ? e : 0.2f * e;
      if (e > m) { s = s * __expf(m - e) + 1.0f; m = e; }
      else       { s += __expf(e - m); }
    }
  }
  for (int off = 16; off; off >>= 1) {
    float m2 = __shfl_xor(m, off, 32);
    float s2 = __shfl_xor(s, off, 32);
    float M  = fmaxf(m, m2);
    s = s * __expf(m - M) + s2 * __expf(m2 - M);
    m = M;
  }
  if (lane == 0) { sm[wv] = m; ss[wv] = s; }
  __syncthreads();
  if (threadIdx.x == 0) {
    float M = sm[0], S = ss[0];
    for (int w = 1; w < 8; ++w) {
      float Mn = fmaxf(M, sm[w]);
      S = S * __expf(M - Mn) + ss[w] * __expf(sm[w] - Mn);
      M = Mn;
    }
    Mrow[row] = M;
    invZ[row] = (S > 0.0f) ? (1.0f / S) : 0.0f;
  }
}

// ---------------------------------------------------------------------------
// TDM helper: DMA one contiguous 16 KB B-slab (4096 dwords) into LDS at byte
// offset `ldsByteOff`. D# per CDNA5 ISA:
//   g0: count=1 | lds_addr | global_addr[56:0] | type=2
//   g1: data_size=4B, tile = 4096 x 1 dwords, tensor_dim0 = 1M dwords (no OOB)
// Whole-wave instruction (EXEC ignored); tracked by TENSORcnt.
// This toolchain's builtin is the 6-arg form: (v4u, v8i, v4i, v4i, v8i, cpol).
// ---------------------------------------------------------------------------
#ifdef HAVE_TDM
__device__ __forceinline__ void tdm_stage(const unsigned int* src,
                                          unsigned int ldsByteOff) {
  unsigned long long ga = (unsigned long long)(uintptr_t)src;
  v4u g0;
  g0.x = 1u;                                                   // count=1
  g0.y = ldsByteOff;                                           // lds_addr
  g0.z = (unsigned int)ga;                                     // global_addr lo
  g0.w = (unsigned int)((ga >> 32) & 0x1FFFFFFull) | (2u << 30); // hi | type=2
  v8i g1;
  g1[0] = (int)(2u << 16);          // workgroup_mask=0, data_size=4B, no flags
  g1[1] = 0;                        // tensor_dim0[15:0] = 0 (in bits 31:16)
  g1[2] = (int)(0x10u | (1u << 16)); // tensor_dim0 = 0x100000 dwords; dim1 = 1
  g1[3] = (int)(4096u << 16);       // tile_dim0 = 4096 dwords (16 KB)
  g1[4] = 1;                        // tile_dim1 = 1, tile_dim2 = 0
  g1[5] = 4096;                     // tensor_dim0_stride = 4096 dwords
  g1[6] = 0;
  g1[7] = 0;
  v4i g2 = {0, 0, 0, 0};
  v4i g3 = {0, 0, 0, 0};
  v8i g4 = {0, 0, 0, 0, 0, 0, 0, 0};
  __builtin_amdgcn_tensor_load_to_lds(g0, g1, g2, g3, g4, 0);
}
#endif

// ---------------------------------------------------------------------------
// Kernel 5: fused softmax-weights x Wh GEMM via V_WMMA_F32_16X16X32_BF16.
// 8 waves/block, wave owns 16 rows x 256 cols (16 accumulator tiles).
// Double-buffered LDS B-slabs fed by the Tensor Data Mover: wave 0 issues
// tensor_load_to_lds for step+1, all waves compute on the current slab, then
// s_wait_tensorcnt 0 + barrier flips buffers. j split over gridDim.y=8 with
// fp32 atomics combining partials.
// ---------------------------------------------------------------------------
__global__ void __launch_bounds__(256, 1)
attn_gemm(const unsigned int* __restrict__ staged,
          const unsigned int* __restrict__ packed,
          const float* __restrict__ fs, const float* __restrict__ fd,
          const float* __restrict__ Mrow, const float* __restrict__ invZ,
          float* __restrict__ hp) {
  // [0, 8192): two 16 KB B slabs; [8192, 8256): two 32-float fd chunks.
  // Single __shared__ object => LDS offset 0, so TDM lds_addr = buf*16384.
  __shared__ __align__(16) unsigned int ldsAll[8256];
  float* ldsF = (float*)ldsAll;

  int lane = threadIdx.x & 31, wv = threadIdx.x >> 5;
  int rowBase = (blockIdx.x * 8 + wv) * 16;
  int ml = lane & 15, hi = lane >> 4;
  int row = rowBase + ml;                 // A-matrix row this lane produces
  float fsr = fs[row], Mm = Mrow[row], iZ = invZ[row];

  v8f acc[16];
#pragma unroll
  for (int t = 0; t < 16; ++t)
#pragma unroll
    for (int r = 0; r < 8; ++r) acc[t][r] = 0.0f;

  int jt0 = blockIdx.y * 32;              // 8 chunks x 32 steps x K=32 = 8192

  // prologue: stage buffer 0
#ifdef HAVE_TDM
  if (wv == 0) tdm_stage(staged + jt0 * 4096, 0u);
#else
  {
    const uint4* s = (const uint4*)(staged + jt0 * 4096);
    uint4* d = (uint4*)ldsAll;
    for (int q = threadIdx.x; q < 1024; q += 256) d[q] = s[q];
  }
#endif
  if (threadIdx.x < 32) ldsF[8192 + threadIdx.x] = fd[jt0 * 32 + threadIdx.x];
  TDM_WAIT();
  __syncthreads();

  for (int step = 0; step < 32; ++step) {
    int jt  = jt0 + step;
    int cur = step & 1, nxt = cur ^ 1;

    // kick off DMA of the next slab while we compute on the current one
    if (step < 31) {
#ifdef HAVE_TDM
      if (wv == 0) tdm_stage(staged + (jt + 1) * 4096, (unsigned)nxt * 16384u);
#else
      const uint4* s = (const uint4*)(staged + (jt + 1) * 4096);
      uint4* d = (uint4*)(ldsAll + nxt * 4096);
      for (int q = threadIdx.x; q < 1024; q += 256) d[q] = s[q];
#endif
      if (threadIdx.x < 32)
        ldsF[8192 + nxt * 32 + threadIdx.x] = fd[(jt + 1) * 32 + threadIdx.x];
      __builtin_prefetch(&packed[row * 256 + jt + 1], 0, 1);
    }

    // build A = P[rowBase:+16, jt*32:+32] in bf16 (documented 16x32 A layout)
    unsigned int w = packed[row * 256 + jt];
    v16bf a;
#pragma unroll
    for (int vg = 0; vg < 8; ++vg) {
#pragma unroll
      for (int h = 0; h < 2; ++h) {
        int k = 2 * (vg & 3) + h + 8 * hi + 16 * (vg >> 2);
        float e = fsr + ldsF[8192 + cur * 32 + k];
        e = e > 0.0f ? e : 0.2f * e;
        float p = ((w >> k) & 1u) ? __expf(e - Mm) * iZ : 0.0f;
        a[2 * vg + h] = (__bf16)p;
      }
    }

#pragma unroll
    for (int nt = 0; nt < 16; ++nt) {
      const v16bf* bp = (const v16bf*)(ldsAll + cur * 4096 + (nt * 32 + lane) * 8);
      v16bf b = *bp;                      // two ds_load_b128
      acc[nt] = __builtin_amdgcn_wmma_f32_16x16x32_bf16(false, a, false, b,
                                                        (short)0, acc[nt],
                                                        false, false);
    }

    TDM_WAIT();          // non-issuing waves have TENSORcnt==0: free
    __syncthreads();
  }

  // combine j-chunks: C layout row = rowBase + r + 8*hi, col = nt*16 + ml
#pragma unroll
  for (int nt = 0; nt < 16; ++nt)
#pragma unroll
    for (int r = 0; r < 8; ++r)
      atomicAdd(&hp[(rowBase + r + 8 * hi) * DD + nt * 16 + ml], acc[nt][r]);
}

// ---------------------------------------------------------------------------
// Kernel 6: ELU epilogue.
// ---------------------------------------------------------------------------
__global__ void elu_kernel(const float* __restrict__ hp, float* __restrict__ out, int n) {
  int i = blockIdx.x * 256 + threadIdx.x;
  if (i < n) {
    float v = hp[i];
    out[i] = v > 0.0f ? v : (__expf(v) - 1.0f);
  }
}

// ---------------------------------------------------------------------------
extern "C" void kernel_launch(void* const* d_in, const int* in_sizes, int n_in,
                              void* d_out, int out_size, void* d_ws, size_t ws_size,
                              hipStream_t stream) {
  const float* X     = (const float*)d_in[0];   // [8192,256]
  const int*   adj   = (const int*)d_in[1];     // [8192,8192]
  const float* W     = (const float*)d_in[2];   // [256,256]
  const float* a_src = (const float*)d_in[3];   // [256]
  const float* a_dst = (const float*)d_in[4];   // [256]
  float* out = (float*)d_out;                   // [8192,256]

  char* ws = (char*)d_ws;
  float*        Wh     = (float*)(ws);                          //  8 MB
  unsigned int* staged = (unsigned int*)(ws + (8u  << 20));     //  4 MB
  float*        fs     = (float*)(ws + (12u << 20));            // 32 KB
  float*        fdv    = fs + NN;
  float*        Mrow   = fdv + NN;
  float*        invZ   = Mrow + NN;
  unsigned int* packed = (unsigned int*)(ws + (13u << 20));     //  8 MB
  float*        hp     = (float*)(ws + (21u << 20));            //  8 MB

  zero_kernel    <<<NN, 256, 0, stream>>>(hp, NN * DD);
  wh_gemm        <<<1024, 256, 0, stream>>>(X, W, Wh);
  fvec_kernel    <<<1024, 256, 0, stream>>>(Wh, a_src, a_dst, fs, fdv);
  repack_kernel  <<<4096, 256, 0, stream>>>(Wh, staged);
  rowstats_kernel<<<NN, 256, 0, stream>>>(adj, fs, fdv, Mrow, invZ, packed);
  attn_gemm      <<<dim3(64, 8), 256, 0, stream>>>(staged, packed, fs, fdv, Mrow, invZ, hp);
  elu_kernel     <<<NN, 256, 0, stream>>>(hp, out, NN * DD);
}
